// VAE_12481174962949
// MI455X (gfx1250) — matrix operand order repaired and checked by reading, not compiled, not run on hardware
//
#include <hip/hip_runtime.h>
#include <hip/hip_bf16.h>

typedef _Float16 f16;
typedef __attribute__((ext_vector_type(16))) _Float16 v16h;
typedef __attribute__((ext_vector_type(8)))  _Float16 v8h;
typedef __attribute__((ext_vector_type(8)))  float    v8f;

#define LAT   64
#define BATCH 256
#define NEMB  4096   // 64*64 SOM nodes

__device__ __forceinline__ float lrelu(float v) { return v > 0.f ? v : 0.01f * v; }

// ---------------- encoder: x -> h10 -> h50 -> (mu, logvar) -> z_e ----------------
__global__ void k_encoder(const float* __restrict__ x,
                          const float* __restrict__ eps,
                          const float* __restrict__ w_e0, const float* __restrict__ b_e0,
                          const float* __restrict__ w_e1, const float* __restrict__ b_e1,
                          const float* __restrict__ w_mu, const float* __restrict__ b_mu,
                          const float* __restrict__ w_lv, const float* __restrict__ b_lv,
                          float* __restrict__ out_ze,
                          float* __restrict__ ws_ze_f32,
                          f16*   __restrict__ ws_ze_f16) {
  int b = blockIdx.x * blockDim.x + threadIdx.x;
  if (b >= BATCH) return;
  float xv = x[b];
  float h0[10];
  for (int j = 0; j < 10; ++j) h0[j] = lrelu(xv * w_e0[j] + b_e0[j]);
  float h1[50];
  for (int j = 0; j < 50; ++j) {
    float s = b_e1[j];
    for (int k = 0; k < 10; ++k) s += h0[k] * w_e1[k * 50 + j];
    h1[j] = lrelu(s);
  }
  for (int j = 0; j < LAT; ++j) {
    float mu = b_mu[j], lv = b_lv[j];
    for (int k = 0; k < 50; ++k) {
      mu += h1[k] * w_mu[k * LAT + j];
      lv += h1[k] * w_lv[k * LAT + j];
    }
    float z = mu + eps[b * LAT + j] * __expf(0.5f * lv);
    out_ze[b * LAT + j]    = z;
    ws_ze_f32[b * LAT + j] = z;
    ws_ze_f16[b * LAT + j] = (f16)z;
  }
}

// ---------------- prep: embeddings f32 -> f16, plus ||e_n||^2 ----------------
__global__ void k_prep_emb(const float* __restrict__ emb,
                           f16* __restrict__ emb16,
                           float* __restrict__ norms) {
  int r = blockIdx.x * blockDim.x + threadIdx.x;
  if (r >= NEMB) return;
  float s = 0.f;
  for (int k = 0; k < LAT; ++k) {
    float v = emb[r * LAT + k];
    s += v * v;
    emb16[r * LAT + k] = (f16)v;
  }
  norms[r] = s;
}

// ---------------- WMMA distance + fused argmin ----------------
// score[b][n] = ||e_n||^2 - 2 * dot(z_e[b], e_n)   (||z||^2 constant per row -> dropped)
// One wave per 16-row M tile. 2-way software pipeline over N tiles: two
// independent accumulators so tile B's WMMAs overlap tile A's argmin VALU,
// filling the WMMA->VALU hazard slots (4 co-exec NOPs for F16 WMMA).
__global__ void __launch_bounds__(256)
k_dist_argmin(const f16* __restrict__ ze16,
              const f16* __restrict__ emb16,
              const float* __restrict__ norms,
              int* __restrict__ best_idx) {
  __shared__ float s_val[8][16][16];
  __shared__ int   s_idx[8][16][16];

  const int tid  = threadIdx.x;
  const int wave = tid >> 5;          // 0..7
  const int lane = tid & 31;
  const int col  = lane & 15;         // N (for B/C) or M (for A) within tile
  const int hi   = lane >> 4;         // which K half-group this lane holds
  const int m0   = (blockIdx.x * 8 + wave) * 16;   // 16 waves total -> 256 rows

  // A fragments (z_e rows), contiguous K layout matching B:
  // lanes 0-15 hold K 0..15(+off), lanes 16-31 hold K 16..31(+off)
  const f16* arow = ze16 + (m0 + col) * LAT;
  v16h a0, a1;
  {
    const int base = hi ? 16 : 0;
    const v8h lo0 = *(const v8h*)(arow + base);
    const v8h hi0 = *(const v8h*)(arow + base + 8);
    const v8h lo1 = *(const v8h*)(arow + base + 32);
    const v8h hi1 = *(const v8h*)(arow + base + 40);
    for (int e = 0; e < 8; ++e) {
      a0[e] = lo0[e]; a0[8 + e] = hi0[e];
      a1[e] = lo1[e]; a1[8 + e] = hi1[e];
    }
  }

  float best[8];
  int   bidx[8];
  for (int r = 0; r < 8; ++r) { best[r] = 3.4e38f; bidx[r] = 0; }

  const int boff = hi ? 16 : 0;

  for (int nt = 0; nt < NEMB / 16; nt += 2) {
    const int nA = nt * 16 + col;
    const int nB = nA + 16;
    // B fragments (e_n as column): lanes 0-15 K=0..15, lanes 16-31 K=16..31
    const f16* browA = emb16 + nA * LAT + boff;
    const f16* browB = emb16 + nB * LAT + boff;
    v16h bA0 = *(const v16h*)(browA);
    v16h bA1 = *(const v16h*)(browA + 32);
    v16h bB0 = *(const v16h*)(browB);
    v16h bB1 = *(const v16h*)(browB + 32);
    if (nt + 2 < NEMB / 16)
      __builtin_prefetch(emb16 + (nA + 32) * LAT, 0, 1);  // global_prefetch_b8

    v8f cA = {}, cB = {};
    cA = __builtin_amdgcn_wmma_f32_16x16x32_f16(false, a0, false, bA0,
                                                (short)0, cA, false, false);
    cB = __builtin_amdgcn_wmma_f32_16x16x32_f16(false, a0, false, bB0,
                                                (short)0, cB, false, false);
    cA = __builtin_amdgcn_wmma_f32_16x16x32_f16(false, a1, false, bA1,
                                                (short)0, cA, false, false);
    cB = __builtin_amdgcn_wmma_f32_16x16x32_f16(false, a1, false, bB1,
                                                (short)0, cB, false, false);

    const float nnA = norms[nA];
    const float nnB = norms[nB];
    for (int r = 0; r < 8; ++r) {
      float sA = nnA - 2.0f * cA[r];
      if (sA < best[r]) { best[r] = sA; bidx[r] = nA; }
    }
    for (int r = 0; r < 8; ++r) {
      float sB = nnB - 2.0f * cB[r];
      if (sB < best[r]) { best[r] = sB; bidx[r] = nB; }
    }
  }

  // C layout: element (r, lane) -> M = r + 8*(lane/16), N = lane%16
  for (int r = 0; r < 8; ++r) {
    int m = r + 8 * hi;
    s_val[wave][m][col] = best[r];
    s_idx[wave][m][col] = bidx[r];
  }
  __syncthreads();

  if (tid < 128) {
    int w = tid >> 4, m = tid & 15;
    float bv = s_val[w][m][0];
    int   bi = s_idx[w][m][0];
    for (int c2 = 1; c2 < 16; ++c2) {
      float v = s_val[w][m][c2];
      if (v < bv) { bv = v; bi = s_idx[w][m][c2]; }
    }
    best_idx[(blockIdx.x * 8 + w) * 16 + m] = bi;
  }
}

// ---------------- gather z_q + neighbors (incl. faithful z_q_right == 0 bug) ----------------
__global__ void k_gather(const float* __restrict__ emb,
                         const int* __restrict__ best_idx,
                         float* __restrict__ out_zq,
                         float* __restrict__ out_nb,
                         float* __restrict__ ws_zq) {
  int g = blockIdx.x * blockDim.x + threadIdx.x;
  if (g >= BATCH * LAT) return;
  int b = g / LAT, j = g % LAT;
  int n  = best_idx[b];
  int nx = n >> 6, ny = n & 63;
  float zq = emb[n * LAT + j];
  out_zq[b * LAT + j] = zq;
  ws_zq[b * LAT + j]  = zq;
  float* nb = out_nb + b * 5 * LAT;
  nb[0 * LAT + j] = zq;
  nb[1 * LAT + j] = (nx < 63) ? emb[((nx + 1) * 64 + ny) * LAT + j] : 0.f;  // up
  nb[2 * LAT + j] = (nx > 0)  ? emb[((nx - 1) * 64 + ny) * LAT + j] : 0.f;  // down
  nb[3 * LAT + j] = 0.f;                                                    // right (== bug)
  nb[4 * LAT + j] = (ny > 0)  ? emb[(nx * 64 + ny - 1) * LAT + j] : 0.f;    // left
}

// ---------------- decoder: 64 -> 100 -> 60 -> 30 -> 1 (lrelu everywhere) ----------------
__global__ void k_decoder(const float* __restrict__ ws_ze,
                          const float* __restrict__ ws_zq,
                          const float* __restrict__ w_d,  const float* __restrict__ b_d,
                          const float* __restrict__ w_d0, const float* __restrict__ b_d0,
                          const float* __restrict__ w_d1, const float* __restrict__ b_d1,
                          const float* __restrict__ w_d2, const float* __restrict__ b_d2,
                          float* __restrict__ out_de,
                          float* __restrict__ out_dq) {
  int g = blockIdx.x * blockDim.x + threadIdx.x;
  if (g >= 2 * BATCH) return;
  int which = g >> 8;       // 0: decode z_e, 1: decode z_q
  int b = g & 255;
  const float* z = (which ? ws_zq : ws_ze) + b * LAT;
  float t0[100];
  for (int j = 0; j < 100; ++j) {
    float s = b_d[j];
    for (int k = 0; k < LAT; ++k) s += z[k] * w_d[k * 100 + j];
    t0[j] = lrelu(s);
  }
  float t1[60];
  for (int j = 0; j < 60; ++j) {
    float s = b_d0[j];
    for (int k = 0; k < 100; ++k) s += t0[k] * w_d0[k * 60 + j];
    t1[j] = lrelu(s);
  }
  float t2[30];
  for (int j = 0; j < 30; ++j) {
    float s = b_d1[j];
    for (int k = 0; k < 60; ++k) s += t1[k] * w_d1[k * 30 + j];
    t2[j] = lrelu(s);
  }
  float s = b_d2[0];
  for (int k = 0; k < 30; ++k) s += t2[k] * w_d2[k];
  s = lrelu(s);
  (which ? out_dq : out_de)[b] = s;
}

extern "C" void kernel_launch(void* const* d_in, const int* in_sizes, int n_in,
                              void* d_out, int out_size, void* d_ws, size_t ws_size,
                              hipStream_t stream) {
  // inputs in setup_inputs() order
  const float* x    = (const float*)d_in[0];
  const float* eps  = (const float*)d_in[1];
  const float* emb  = (const float*)d_in[2];
  const float* w_e0 = (const float*)d_in[3];
  const float* b_e0 = (const float*)d_in[4];
  const float* w_e1 = (const float*)d_in[5];
  const float* b_e1 = (const float*)d_in[6];
  const float* w_mu = (const float*)d_in[7];
  const float* b_mu = (const float*)d_in[8];
  const float* w_lv = (const float*)d_in[9];
  const float* b_lv = (const float*)d_in[10];
  const float* w_d  = (const float*)d_in[11];
  const float* b_d  = (const float*)d_in[12];
  const float* w_d0 = (const float*)d_in[13];
  const float* b_d0 = (const float*)d_in[14];
  const float* w_d1 = (const float*)d_in[15];
  const float* b_d1 = (const float*)d_in[16];
  const float* w_d2 = (const float*)d_in[17];
  const float* b_d2 = (const float*)d_in[18];

  float* out = (float*)d_out;
  float* out_ze = out;                       // [256,64]
  float* out_zq = out + 16384;               // [256,64]
  float* out_nb = out + 32768;               // [256,5,64]
  float* out_de = out + 114688;              // [256]
  float* out_dq = out + 114944;              // [256]

  // workspace layout (byte offsets, all >=256B aligned)
  char* ws = (char*)d_ws;
  f16*   ws_emb16 = (f16*)(ws + 0);          // 4096*64*2 = 524288
  float* ws_norms = (float*)(ws + 524288);   // 4096*4   =  16384
  float* ws_ze32  = (float*)(ws + 540672);   // 256*64*4 =  65536
  f16*   ws_ze16  = (f16*)(ws + 606208);     // 256*64*2 =  32768
  int*   ws_best  = (int*)(ws + 638976);     // 256*4    =   1024
  float* ws_zq32  = (float*)(ws + 640000);   // 256*64*4 =  65536

  k_prep_emb<<<16, 256, 0, stream>>>(emb, ws_emb16, ws_norms);
  k_encoder<<<1, 256, 0, stream>>>(x, eps, w_e0, b_e0, w_e1, b_e1,
                                   w_mu, b_mu, w_lv, b_lv,
                                   out_ze, ws_ze32, ws_ze16);
  k_dist_argmin<<<2, 256, 0, stream>>>(ws_ze16, ws_emb16, ws_norms, ws_best);
  k_gather<<<64, 256, 0, stream>>>(emb, ws_best, out_zq, out_nb, ws_zq32);
  k_decoder<<<2, 256, 0, stream>>>(ws_ze32, ws_zq32,
                                   w_d, b_d, w_d0, b_d0, w_d1, b_d1, w_d2, b_d2,
                                   out_de, out_dq);
}
